// Mesh_To_Grid_Decoder_24996709663141
// MI455X (gfx1250) — matrix operand order repaired and checked by reading, not compiled, not run on hardware
//
#include <hip/hip_runtime.h>

// ---------------------------------------------------------------------------
// Problem constants (from reference setup_inputs)
// ---------------------------------------------------------------------------
constexpr int B_  = 8;
constexpr int F_  = 64;          // feature channels
constexpr int H_  = 128;
constexpr int W_  = 128;
constexpr int V_  = H_ * W_;     // 16384 vertices
constexpr int E_  = 3 * V_;      // 49152 edges -> 2E = 98304 endpoints
constexpr int KIN = 192;         // 6 * F/2
constexpr int MID = 96;
constexpr int OUT = 64;

constexpr int MT  = 128;         // rows (vertices) per block
constexpr int XS  = 200;         // LDS stride for x tile (bf16 elems), conflict-padded
constexpr int YS  = 104;         // LDS stride for y tile (bf16 elems), conflict-padded

typedef __attribute__((ext_vector_type(16))) __bf16 v16bf;
typedef __attribute__((ext_vector_type(8)))  float  v8f;

union FragU {
    v16bf        v;
    unsigned int u[8];
};

__device__ __forceinline__ unsigned short f2bf(float f) {
    unsigned int u = __builtin_bit_cast(unsigned int, f);
    u += 0x7FFFu + ((u >> 16) & 1u);          // round-to-nearest-even
    return (unsigned short)(u >> 16);
}

// ---------------------------------------------------------------------------
// Prep kernels
// ---------------------------------------------------------------------------
__global__ void k_zero_cnt(int* cnt) {
    int i = blockIdx.x * blockDim.x + threadIdx.x;
    if (i < V_) cnt[i] = 0;
}

// endpoint i -> ticket slot of its vertex (arbitrary order; sorted next)
__global__ void k_ticket(const int* __restrict__ edges, int* cnt, int* idx6) {
    int i = blockIdx.x * blockDim.x + threadIdx.x;
    if (i < 2 * E_) {
        int v = edges[i];
        int t = atomicAdd(&cnt[v], 1);
        idx6[v * 6 + t] = i;
    }
}

// sort the 6 endpoint indices per vertex ascending == occurrence-rank slots.
// Also makes idx6 deterministic regardless of atomic ordering.
__global__ void k_sort6(int* idx6) {
    int v = blockIdx.x * blockDim.x + threadIdx.x;
    if (v >= V_) return;
    int a[6];
#pragma unroll
    for (int s = 0; s < 6; ++s) a[s] = idx6[v * 6 + s];
#pragma unroll
    for (int i = 1; i < 6; ++i) {
        int key = a[i];
        int j = i - 1;
        while (j >= 0 && a[j] > key) { a[j + 1] = a[j]; --j; }
        a[j + 1] = key;
    }
#pragma unroll
    for (int s = 0; s < 6; ++s) idx6[v * 6 + s] = a[s];
}

// W1[k,n] (192x96) -> w1t[n,k] bf16 ; W2[k,n] (96x64) -> w2t[n,k] bf16
__global__ void k_cvt_weights(const float* __restrict__ W1,
                              const float* __restrict__ W2,
                              unsigned short* __restrict__ w1t,
                              unsigned short* __restrict__ w2t) {
    int t = blockIdx.x * blockDim.x + threadIdx.x;
    if (t < KIN * MID) {
        int n = t / KIN, k = t - n * KIN;
        w1t[n * KIN + k] = f2bf(W1[k * MID + n]);
    } else if (t < KIN * MID + MID * OUT) {
        int t2 = t - KIN * MID;
        int n = t2 / MID, k = t2 - n * MID;
        w2t[n * MID + k] = f2bf(W2[k * OUT + n]);
    }
}

// ---------------------------------------------------------------------------
// Fused gather + (192->96 ReLU) + (96->64 ReLU) kernel.
// grid = B * (V/MT) = 1024 blocks, 256 threads (8 wave32 waves).
// ---------------------------------------------------------------------------
__global__ void __launch_bounds__(256)
k_fused(const float* __restrict__ features,
        const int* __restrict__ idx6,
        const unsigned short* __restrict__ w1t,
        const unsigned short* __restrict__ w2t,
        const float* __restrict__ b1,
        const float* __restrict__ b2,
        float* __restrict__ out) {
    __shared__ unsigned short xbuf[MT * XS];   // 51200 B; reused for y tile

    const int tid   = threadIdx.x;
    const int b     = blockIdx.x >> 7;           // V/MT = 128 tiles per batch
    const int vbase = (blockIdx.x & 127) * MT;

    // ---- Gather: x[m][s*32+f] = features[b, p*32+f, e], (e,p) from sorted idx6
    for (int el = tid; el < MT * KIN; el += 256) {
        int m = el / KIN;
        int j = el - m * KIN;
        int s = j >> 5, f = j & 31;
        int i = idx6[(vbase + m) * 6 + s];
        int e = i >> 1, p = i & 1;
        float val = features[(size_t)b * F_ * E_ + (size_t)((p << 5) + f) * E_ + e];
        xbuf[m * XS + j] = f2bf(val);
    }
    __syncthreads();

    const int wave  = tid >> 5;
    const int lane  = tid & 31;
    const int lm    = lane & 15;
    const int hk    = (lane < 16) ? 0 : 8;    // A-frag hi-lane K offset == C-frag M offset
    const int bhk   = (lane < 16) ? 0 : 16;   // unused; B frag uses hk (same k order as A)
    const int mbase = wave * 16;

    const unsigned int* w1t32 = (const unsigned int*)w1t;
    const unsigned int* w2t32 = (const unsigned int*)w2t;

    // ---- Layer 1: [128x192] x [192x96] -> [128x96], per wave: 16 rows
    v8f c1[6] = {};
#pragma unroll
    for (int kt = 0; kt < 6; ++kt) {
        const int kb = kt * 32;
        FragU a;
#pragma unroll
        for (int j = 0; j < 8; ++j) {
            int k = kb + ((j & 4) ? 16 : 0) + hk + (j & 3) * 2;
            a.u[j] = *reinterpret_cast<const unsigned int*>(&xbuf[(mbase + lm) * XS + k]);
        }
#pragma unroll
        for (int nt = 0; nt < 6; ++nt) {
            const int n = nt * 16 + lm;
            FragU bf;
#pragma unroll
            for (int j = 0; j < 8; ++j) {
                int k = kb + ((j & 4) ? 16 : 0) + hk + (j & 3) * 2;
                bf.u[j] = w1t32[(n * KIN + k) >> 1];
            }
            c1[nt] = __builtin_amdgcn_wmma_f32_16x16x32_bf16(
                false, a.v, false, bf.v, (short)0, c1[nt], false, false);
        }
    }

    // all waves done reading x before y overwrites the same LDS buffer
    __syncthreads();

    // ---- bias + ReLU + store y tile (bf16) into LDS
#pragma unroll
    for (int nt = 0; nt < 6; ++nt) {
        const int n = nt * 16 + lm;
        const float bias = b1[n];
#pragma unroll
        for (int r = 0; r < 8; ++r) {
            int m = r + hk;                       // C layout: lanes16-31 hold M=8..15
            float vv = c1[nt][r] + bias;
            vv = vv > 0.0f ? vv : 0.0f;
            xbuf[(mbase + m) * YS + n] = f2bf(vv);
        }
    }
    __syncthreads();

    // ---- Layer 2: [128x96] x [96x64] -> [128x64]
    v8f c2[4] = {};
#pragma unroll
    for (int kt = 0; kt < 3; ++kt) {
        const int kb = kt * 32;
        FragU a;
#pragma unroll
        for (int j = 0; j < 8; ++j) {
            int k = kb + ((j & 4) ? 16 : 0) + hk + (j & 3) * 2;
            a.u[j] = *reinterpret_cast<const unsigned int*>(&xbuf[(mbase + lm) * YS + k]);
        }
#pragma unroll
        for (int nt = 0; nt < 4; ++nt) {
            const int n = nt * 16 + lm;
            FragU bf;
#pragma unroll
            for (int j = 0; j < 8; ++j) {
                int k = kb + ((j & 4) ? 16 : 0) + hk + (j & 3) * 2;
                bf.u[j] = w2t32[(n * MID + k) >> 1];
            }
            c2[nt] = __builtin_amdgcn_wmma_f32_16x16x32_bf16(
                false, a.v, false, bf.v, (short)0, c2[nt], false, false);
        }
    }

    // ---- bias + ReLU + store out[b, o, v] (f32)
#pragma unroll
    for (int nt = 0; nt < 4; ++nt) {
        const int o = nt * 16 + lm;
        const float bias = b2[o];
#pragma unroll
        for (int r = 0; r < 8; ++r) {
            int m = r + hk;
            int v = vbase + mbase + m;
            float vv = c2[nt][r] + bias;
            vv = vv > 0.0f ? vv : 0.0f;
            out[(size_t)b * OUT * V_ + (size_t)o * V_ + v] = vv;
        }
    }
}

// ---------------------------------------------------------------------------
// Launcher
// ---------------------------------------------------------------------------
extern "C" void kernel_launch(void* const* d_in, const int* in_sizes, int n_in,
                              void* d_out, int out_size, void* d_ws, size_t ws_size,
                              hipStream_t stream) {
    const float* features = (const float*)d_in[0];
    const int*   conn     = (const int*)d_in[1];
    // d_in[2] = output_dim (H, W, C_out) - hardcoded
    const float* W1 = (const float*)d_in[3];
    const float* b1 = (const float*)d_in[4];
    const float* W2 = (const float*)d_in[5];
    const float* b2 = (const float*)d_in[6];
    float* out = (float*)d_out;

    char* ws = (char*)d_ws;
    int*            cnt  = (int*)ws;                               // V ints
    int*            idx6 = (int*)(ws + (size_t)V_ * 4);            // 6V ints
    unsigned short* w1t  = (unsigned short*)(ws + (size_t)V_ * 4 + (size_t)6 * V_ * 4);
    unsigned short* w2t  = (unsigned short*)((char*)w1t + (size_t)KIN * MID * 2);

    k_zero_cnt<<<(V_ + 255) / 256, 256, 0, stream>>>(cnt);
    k_ticket<<<(2 * E_ + 255) / 256, 256, 0, stream>>>(conn, cnt, idx6);
    k_sort6<<<(V_ + 255) / 256, 256, 0, stream>>>(idx6);
    k_cvt_weights<<<(KIN * MID + MID * OUT + 255) / 256, 256, 0, stream>>>(W1, W2, w1t, w2t);

    k_fused<<<B_ * (V_ / MT), 256, 0, stream>>>(features, idx6, w1t, w2t, b1, b2, out);
}
